// AttentionHead_6906307412546
// MI455X (gfx1250) — matrix operand order repaired and checked by reading, not compiled, not run on hardware
//
#include <hip/hip_runtime.h>
#include <hip/hip_bf16.h>
#include <math.h>

typedef __bf16 bf16;
typedef __attribute__((ext_vector_type(16))) __bf16 v16bf;
typedef __attribute__((ext_vector_type(8)))  __bf16 v8bf;
typedef __attribute__((ext_vector_type(8)))  float  v8f;
typedef __attribute__((ext_vector_type(4)))  float  v4f;
typedef __attribute__((ext_vector_type(4)))  int    v4i;

#define NB     4
#define SEQ    4096
#define DVEC   1024
#define DHEAD  256

#define WMMA_BF16(A, B, C) \
  __builtin_amdgcn_wmma_f32_16x16x32_bf16(false, (A), false, (B), (short)0, (C), false, false)

#ifndef __has_builtin
#define __has_builtin(x) 0
#endif
#if __has_builtin(__builtin_amdgcn_global_load_async_to_lds_b128)
#define HAS_ASYNC_LDS 1
#else
#define HAS_ASYNC_LDS 0
#endif

__device__ __forceinline__ v8f zero_v8f() {
  v8f z;
#pragma unroll
  for (int i = 0; i < 8; ++i) z[i] = 0.0f;
  return z;
}

// 16-byte global -> LDS copy; async (ASYNCcnt) when available.
// Builtin prototype (from compiler diagnostic): (v4i AS1*, v4i AS3*, imm, imm).
__device__ __forceinline__ void copy16_async(bf16* l, const bf16* g) {
#if HAS_ASYNC_LDS
  __builtin_amdgcn_global_load_async_to_lds_b128(
      (__attribute__((address_space(1))) v4i*)(g),
      (__attribute__((address_space(3))) v4i*)(l), 0, 0);
#else
  *(v4i*)l = *(const v4i*)g;
#endif
}

__device__ __forceinline__ void await_async() {
#if HAS_ASYNC_LDS
#if __has_builtin(__builtin_amdgcn_s_wait_asynccnt)
  __builtin_amdgcn_s_wait_asynccnt(0);
#else
  asm volatile("s_wait_asynccnt 0" ::: "memory");
#endif
#endif
}

// A-fragment (16x32 bf16, M rows across lanes 0..15, K pattern per CDNA5 ISA:
// lanes<16 hold K {kb..kb+7, kb+16..kb+23} with kb=0; lanes>=16 kb=8).
__device__ __forceinline__ v16bf load_a_bf16(const bf16* base, int ld, int row0, int k0) {
  const int l  = threadIdx.x & 31;
  const int m  = l & 15;
  const int kb = (l < 16) ? 0 : 8;
  const bf16* p = base + (size_t)(row0 + m) * ld + k0 + kb;
  v8bf lo = *(const v8bf*)(p);       // K = kb .. kb+7
  v8bf hi = *(const v8bf*)(p + 16);  // K = kb+16 .. kb+23
  v16bf r;
#pragma unroll
  for (int i = 0; i < 8; ++i) { r[i] = lo[i]; r[i + 8] = hi[i]; }
  return r;
}

// A-fragment from f32 source, converted to bf16 in-register.
__device__ __forceinline__ v16bf load_a_f32(const float* __restrict__ base, int ld,
                                            int row0, int k0) {
  const int l  = threadIdx.x & 31;
  const int m  = l & 15;
  const int kb = (l < 16) ? 0 : 8;
  const float* p = base + (size_t)(row0 + m) * ld + k0 + kb;
  v4f a0 = *(const v4f*)(p);
  v4f a1 = *(const v4f*)(p + 4);
  v4f b0 = *(const v4f*)(p + 16);
  v4f b1 = *(const v4f*)(p + 20);
  v16bf r;
#pragma unroll
  for (int i = 0; i < 4; ++i) {
    r[i]      = (bf16)a0[i];
    r[i + 4]  = (bf16)a1[i];
    r[i + 8]  = (bf16)b0[i];
    r[i + 12] = (bf16)b1[i];
  }
  return r;
}

// B-fragment (32x16 bf16) for B[k][n] = base[(n0+n)*ld + k]:
// lane n = l&15, K {kb..kb+7, kb+16..kb+23} with kb = k0 + (l<16 ? 0 : 8).
__device__ __forceinline__ v16bf load_bT_bf16(const bf16* base, int ld, int n0, int k0) {
  const int l  = threadIdx.x & 31;
  const int n  = l & 15;
  const int ks = k0 + ((l < 16) ? 0 : 8);
  const bf16* p = base + (size_t)(n0 + n) * ld + ks;
  v8bf lo = *(const v8bf*)(p);
  v8bf hi = *(const v8bf*)(p + 16);
  v16bf r;
#pragma unroll
  for (int i = 0; i < 8; ++i) { r[i] = lo[i]; r[i + 8] = hi[i]; }
  return r;
}

// ---------------------------------------------------------------------------
// Kernel 1: convert the three weight matrices f32 -> bf16 (stay in L2 forever)
// ---------------------------------------------------------------------------
__global__ void cvt_weights_kernel(const float* __restrict__ wq,
                                   const float* __restrict__ wk,
                                   const float* __restrict__ wv,
                                   bf16* __restrict__ dst) {
  const int n = DHEAD * DVEC;
  int i = blockIdx.x * blockDim.x + threadIdx.x;
  if (i < n) {
    dst[i]         = (bf16)wq[i];
    dst[n + i]     = (bf16)wk[i];
    dst[2 * n + i] = (bf16)wv[i];
  }
}

// ---------------------------------------------------------------------------
// Kernel 2: projection  out = x @ W^T + b  -> bf16 (row-major, or transposed
// [DHEAD][SEQ] per batch for V so attention B-fragments are contiguous).
// Block = 256 threads = 8 waves; each wave does a 16x32 slab of the 16x256 tile.
// ---------------------------------------------------------------------------
__global__ void proj_kernel(const float* __restrict__ x, const bf16* __restrict__ w,
                            const float* __restrict__ bias, bf16* __restrict__ out,
                            int transpose_out) {
  const int lane = threadIdx.x & 31;
  const int wave = threadIdx.x >> 5;   // 0..7
  const int row0 = blockIdx.x * 16;    // over NB*SEQ rows
  const int n0   = wave * 32;

  v8f acc0 = zero_v8f();
  v8f acc1 = zero_v8f();
#pragma unroll 4
  for (int kc = 0; kc < DVEC; kc += 32) {
    v16bf a  = load_a_f32(x, DVEC, row0, kc);
    v16bf b0 = load_bT_bf16(w, DVEC, n0, kc);       // B[k][h] = W[h][k]
    v16bf b1 = load_bT_bf16(w, DVEC, n0 + 16, kc);
    acc0 = WMMA_BF16(a, b0, acc0);
    acc1 = WMMA_BF16(a, b1, acc1);
  }

  const int h0   = n0 + (lane & 15);
  const int roff = (lane < 16) ? 0 : 8;
  const float bv0 = bias[h0];
  const float bv1 = bias[h0 + 16];

  if (!transpose_out) {
#pragma unroll
    for (int r = 0; r < 8; ++r) {
      const size_t row = (size_t)(row0 + r + roff);
      out[row * DHEAD + h0]      = (bf16)(acc0[r] + bv0);
      out[row * DHEAD + h0 + 16] = (bf16)(acc1[r] + bv1);
    }
  } else {
    // out is [NB][DHEAD][SEQ]; a 16-row tile never crosses a batch boundary.
    const int bidx = row0 / SEQ;
    const int s0   = row0 - bidx * SEQ;
    bf16* ob = out + (size_t)bidx * DHEAD * SEQ;
    v8bf p0, p1;
#pragma unroll
    for (int r = 0; r < 8; ++r) {
      p0[r] = (bf16)(acc0[r] + bv0);
      p1[r] = (bf16)(acc1[r] + bv1);
    }
    *(v8bf*)(ob + (size_t)h0 * SEQ + s0 + roff)        = p0;  // 16B store
    *(v8bf*)(ob + (size_t)(h0 + 16) * SEQ + s0 + roff) = p1;
  }
}

// ---------------------------------------------------------------------------
// Kernel 3: fused attention. Block = 128 threads = 4 waves = 4 query tiles
// sharing one double-buffered K/V stream staged in LDS with async-to-LDS
// copies (4x reduction of K/V cache traffic; copy of block i+1 overlaps WMMA
// of block i). Each wave owns full softmax rows -> no combine step.
// Sᵀ trick: D-fragment of K×Qᵀ (lane=query, vgpr=key) is bit-compatible with
// the A-fragment layout of P for P×V — mask/scale/exp/cvt is pure lane-local.
// Denominator = extra WMMA vs an all-ones B fragment (replicated to all lanes).
// ---------------------------------------------------------------------------
#define KBUF_ELEMS (32 * DHEAD)  // 16KB per buffer
#define VBUF_ELEMS (DHEAD * 32)  // 16KB per buffer

// Stage one 32-key block: K rows [kb..kb+32) x 256, and V^T slab 256 x 32.
__device__ __forceinline__ void stage_block(bf16* lk, bf16* lv,
                                            const bf16* Kg, const bf16* Vg) {
  const int t = threadIdx.x;  // 0..127
#pragma unroll
  for (int i = 0; i < 8; ++i) {       // K: 1024 chunks of 16B
    int idx = i * 128 + t;
    int row = idx >> 5;               // 32 chunks per 512B row
    int off = (idx & 31) * 8;
    copy16_async(lk + row * DHEAD + off, Kg + (size_t)row * DHEAD + off);
  }
#pragma unroll
  for (int i = 0; i < 8; ++i) {       // V^T: 256 rows x 64B
    int idx = i * 128 + t;
    int row = idx >> 2;               // 4 chunks per 64B row
    int off = (idx & 3) * 8;
    copy16_async(lv + row * 32 + off, Vg + (size_t)row * SEQ + off);
  }
}

__global__ void attn_kernel(const bf16* __restrict__ Q, const bf16* __restrict__ K,
                            const bf16* __restrict__ VT, const int* __restrict__ mask,
                            float* __restrict__ out) {
  extern __shared__ bf16 smem[];  // [K0 | K1 | V0 | V1] = 64KB
  bf16* sK0 = smem;
  bf16* sK1 = smem + KBUF_ELEMS;
  bf16* sV0 = smem + 2 * KBUF_ELEMS;
  bf16* sV1 = smem + 2 * KBUF_ELEMS + VBUF_ELEMS;

  const int lane = threadIdx.x & 31;
  const int wave = threadIdx.x >> 5;       // 0..3
  const int tile = blockIdx.x * 4 + wave;  // 0..1023 (all 4 in same batch)
  const int bidx = tile >> 8;              // tile / (SEQ/16)
  const int q0   = (tile & 255) * 16;

  const bf16* Qb = Q  + (size_t)bidx * SEQ * DHEAD;
  const bf16* Kg = K  + (size_t)bidx * SEQ * DHEAD;
  const bf16* Vg = VT + (size_t)bidx * DHEAD * SEQ;
  const int* mrow = mask + (size_t)bidx * SEQ * SEQ + (size_t)(q0 + (lane & 15)) * SEQ;

  // Q tile as 8 fixed B fragments (Bᵀ: lane = query, contiguous head-dim run)
  v16bf bq[8];
#pragma unroll
  for (int kc = 0; kc < 8; ++kc) bq[kc] = load_bT_bf16(Qb, DHEAD, q0, kc * 32);

  v16bf ones;
#pragma unroll
  for (int i = 0; i < 16; ++i) ones[i] = (bf16)1.0f;

  v8f o[16];
#pragma unroll
  for (int h = 0; h < 16; ++h) o[h] = zero_v8f();
  v8f dsum = zero_v8f();

  const int koffm = (lane < 16) ? 0 : 8;

  stage_block(sK0, sV0, Kg, Vg);  // prologue: block 0 into buffer 0

  for (int blk = 0; blk < SEQ / 32; ++blk) {
    const int kb  = blk * 32;
    const int cur = blk & 1;
    bf16* lk = cur ? sK1 : sK0;
    bf16* lv = cur ? sV1 : sV0;

    await_async();      // my async copies into buffer `cur` done
    __syncthreads();    // everyone's copies visible; prev buffer fully consumed

    if (blk + 1 < SEQ / 32) {  // overlap next block's copy with this block's math
      bf16* nk = cur ? sK0 : sK1;
      bf16* nv = cur ? sV0 : sV1;
      stage_block(nk, nv, Kg + (size_t)(kb + 32) * DHEAD, Vg + kb + 32);
    }

    // ---- Sᵀ = K_block(16x256 as A, x2) × Qᵀ(B) : f32 accum, from LDS ----
    v8f s0 = zero_v8f();
    v8f s1 = zero_v8f();
#pragma unroll
    for (int kc = 0; kc < 8; ++kc) {
      v16bf a0 = load_a_bf16(lk, DHEAD, 0,  kc * 32);
      v16bf a1 = load_a_bf16(lk, DHEAD, 16, kc * 32);
      s0 = WMMA_BF16(a0, bq[kc], s0);
      s1 = WMMA_BF16(a1, bq[kc], s1);
    }

    // ---- mask (fill 0 pre-softmax, per reference), scale, exp ----
    v4i m0a = *(const v4i*)(mrow + kb + koffm);
    v4i m0b = *(const v4i*)(mrow + kb + koffm + 4);
    v4i m1a = *(const v4i*)(mrow + kb + 16 + koffm);
    v4i m1b = *(const v4i*)(mrow + kb + 16 + koffm + 4);
    if (kb + 32 < SEQ) __builtin_prefetch(mrow + kb + 32 + koffm, 0, 0);

    v16bf pa;  // P as A-fragment, directly from D layout (no cross-lane moves)
#pragma unroll
    for (int r = 0; r < 4; ++r) {
      float t0 = (m0a[r] == 0) ? 0.0f : s0[r]     * 0.0625f;  // 1/sqrt(256)
      float t1 = (m0b[r] == 0) ? 0.0f : s0[r + 4] * 0.0625f;
      float t2 = (m1a[r] == 0) ? 0.0f : s1[r]     * 0.0625f;
      float t3 = (m1b[r] == 0) ? 0.0f : s1[r + 4] * 0.0625f;
      pa[r]      = (bf16)__expf(t0);   // scores are O(1): max-free softmax exact
      pa[r + 4]  = (bf16)__expf(t1);
      pa[r + 8]  = (bf16)__expf(t2);
      pa[r + 12] = (bf16)__expf(t3);
    }

    // ---- O += P × V ;  denom += P × ones (from LDS) ----
#pragma unroll
    for (int h = 0; h < 16; ++h) {
      v16bf bv = load_bT_bf16(lv, 32, h * 16, 0);
      o[h] = WMMA_BF16(pa, bv, o[h]);
    }
    dsum = WMMA_BF16(pa, ones, dsum);

    __syncthreads();   // all waves done reading `cur` before it is re-staged
  }

  // ---- divide by softmax denominator and store f32 output ----
  const int roff = (lane < 16) ? 0 : 8;
  float inv[8];
#pragma unroll
  for (int r = 0; r < 8; ++r) inv[r] = 1.0f / dsum[r];

  float* ob = out + ((size_t)bidx * SEQ + q0) * DHEAD;
#pragma unroll
  for (int h = 0; h < 16; ++h) {
    const int hh = h * 16 + (lane & 15);
#pragma unroll
    for (int r = 0; r < 8; ++r)
      ob[(size_t)(r + roff) * DHEAD + hh] = o[h][r] * inv[r];
  }
}

// ---------------------------------------------------------------------------
extern "C" void kernel_launch(void* const* d_in, const int* in_sizes, int n_in,
                              void* d_out, int out_size, void* d_ws, size_t ws_size,
                              hipStream_t stream) {
  (void)in_sizes; (void)n_in; (void)out_size; (void)ws_size;

  const float* x_q  = (const float*)d_in[0];
  const float* x_k  = (const float*)d_in[1];
  const float* x_v  = (const float*)d_in[2];
  const int*   mask = (const int*)d_in[3];
  const float* wq_w = (const float*)d_in[4];
  const float* wq_b = (const float*)d_in[5];
  const float* wk_w = (const float*)d_in[6];
  const float* wk_b = (const float*)d_in[7];
  const float* wv_w = (const float*)d_in[8];
  const float* wv_b = (const float*)d_in[9];

  // Workspace layout (bf16): weights 1.5MB, then Q, K, V^T (8MB each) ~ 25.5MB
  char* ws = (char*)d_ws;
  bf16* wbf = (bf16*)ws;
  bf16* Qb  = (bf16*)(ws + (size_t)3 * DHEAD * DVEC * sizeof(bf16));
  bf16* Kb  = Qb + (size_t)NB * SEQ * DHEAD;
  bf16* Vt  = Kb + (size_t)NB * SEQ * DHEAD;

  const int wn = DHEAD * DVEC;  // 262144
  cvt_weights_kernel<<<(wn + 255) / 256, 256, 0, stream>>>(wq_w, wk_w, wv_w, wbf);

  const int rows = NB * SEQ;  // 16384
  proj_kernel<<<rows / 16, 256, 0, stream>>>(x_q, wbf,          wq_b, Qb, 0);
  proj_kernel<<<rows / 16, 256, 0, stream>>>(x_k, wbf + wn,     wk_b, Kb, 0);
  proj_kernel<<<rows / 16, 256, 0, stream>>>(x_v, wbf + 2 * wn, wv_b, Vt, 1);

  const size_t ldsBytes = (2 * KBUF_ELEMS + 2 * VBUF_ELEMS) * sizeof(bf16);  // 64KB
  attn_kernel<<<(rows / 16) / 4, 128, ldsBytes, stream>>>(Qb, Kb, Vt, mask, (float*)d_out);
}
